// DLRM_Net_multi_GPU_46849503265113
// MI455X (gfx1250) — compile-verified, matrix operand order checked by reading, not run on hardware
//
#include <hip/hip_runtime.h>
#include <math.h>
#include <stdint.h>

// ---------- problem constants (from reference) ----------
#define NUM_TABLES 26
#define NUM_ROWSE  100000
#define EMB_DIM    128
#define BATCH      8192
#define BAG        8

typedef __attribute__((ext_vector_type(16))) __bf16 v16bf;
typedef __attribute__((ext_vector_type(8)))  __bf16 bf16x8;
typedef __attribute__((ext_vector_type(4)))  __bf16 bf16x4;
typedef __attribute__((ext_vector_type(8)))  float  v8f;

static __device__ __forceinline__ v16bf zero16bf() {
  v16bf z;
#pragma unroll
  for (int i = 0; i < 16; ++i) z[i] = (__bf16)0.0f;
  return z;
}
static __device__ __forceinline__ v8f zero8f() {
  v8f z;
#pragma unroll
  for (int i = 0; i < 8; ++i) z[i] = 0.0f;
  return z;
}
// Assemble the 16-bf16-per-lane WMMA operand: two contiguous 8-element chunks.
static __device__ __forceinline__ v16bf load_tile_row(const __bf16* p) {
  bf16x8 lo = *(const bf16x8*)(p);
  bf16x8 hi = *(const bf16x8*)(p + 16);
  v16bf a;
#pragma unroll
  for (int i = 0; i < 8; ++i) { a[i] = lo[i]; a[i + 8] = hi[i]; }
  return a;
}
static __device__ __forceinline__ v8f wmma_bf16(v16bf a, v16bf b, v8f c) {
  return __builtin_amdgcn_wmma_f32_16x16x32_bf16(false, a, false, b, (short)0, c,
                                                 false, false);
}

// ---------- fp32 -> bf16 convert with K padding (weights, dense input) ----------
__global__ void cvt_pad_bf16(const float* __restrict__ src, __bf16* __restrict__ dst,
                             int K, int Kpad, long total) {
  long i = (long)blockIdx.x * blockDim.x + threadIdx.x;
  if (i >= total) return;
  int n = (int)(i / Kpad);
  int k = (int)(i % Kpad);
  float v = (k < K) ? src[(long)n * K + k] : 0.0f;
  dst[i] = (__bf16)v;
}

// ---------- embedding bag-sum: one wave per (table, sample) ----------
__global__ void __launch_bounds__(256) embed_bag_sum(
    const int* __restrict__ idx, const float* __restrict__ tables,
    __bf16* __restrict__ T) {
  int wave = blockIdx.x * (blockDim.x >> 5) + (threadIdx.x >> 5);
  int lane = threadIdx.x & 31;
  int t = wave / BATCH;
  int b = wave - t * BATCH;
  const int*   ip  = idx + ((long)t * BATCH + b) * BAG;
  const float* tab = tables + (long)t * NUM_ROWSE * EMB_DIM;
  float4 acc = make_float4(0.f, 0.f, 0.f, 0.f);
#pragma unroll
  for (int j = 0; j < BAG; ++j) {
    long r = (long)ip[j];
    if (j + 1 < BAG)  // global_prefetch of the next bag row
      __builtin_prefetch(tab + (long)ip[j + 1] * EMB_DIM + lane * 4, 0, 1);
    float4 v = ((const float4*)(tab + r * EMB_DIM))[lane];
    acc.x += v.x; acc.y += v.y; acc.z += v.z; acc.w += v.w;
  }
  bf16x4 o;
  o[0] = (__bf16)acc.x; o[1] = (__bf16)acc.y; o[2] = (__bf16)acc.z; o[3] = (__bf16)acc.w;
  *(bf16x4*)(T + (((long)b * 27) + 1 + t) * EMB_DIM + lane * 4) = o;
}

// ---------- bf16 WMMA GEMM: out[m,n] = relu(A[m,:] . W[n,:] + bias[n]) ----------
// A: BATCH x Kpad (bf16), W: N x Kpad (bf16), out: bf16 with row stride ldo.
// Wave computes 32(M) x 64(N); block = 8 waves -> 256(M) x 64(N).
// W tiles (64x32 bf16, 4KB) are staged into LDS with async global->LDS copies,
// double-buffered; A tiles are register double-buffered. All 4 B tiles are
// pulled from LDS in one clause, then 8 WMMAs issue back-to-back.
__global__ void __launch_bounds__(256) gemm_bf16_wmma(
    const __bf16* __restrict__ A, const __bf16* __restrict__ W,
    const float* __restrict__ bias, __bf16* __restrict__ out,
    int Kpad, int ldo) {
  __shared__ __bf16 sW[2][64 * 32];

  int tid  = threadIdx.x;
  int lane = tid & 31;
  int wave = tid >> 5;
  int half = lane >> 4;        // 0: k-chunks {0..7,16..23}; 1: {8..15,24..31}
  int rlow = lane & 15;
  int mBase = (blockIdx.y * 8 + wave) * 32;
  int nBase = blockIdx.x * 64;

  // async staging: each thread owns one 16-byte chunk of the 64x32 W tile
  // (one global_load_async_to_lds_b128 per wave moves 512B)
  int srow   = tid >> 2;         // 0..63 (row of the W tile)
  int schunk = (tid & 3) * 8;    // element offset within the row
  const __bf16* wsrc = W + (long)(nBase + srow) * Kpad + schunk;
  unsigned ldsDst[2] = {
      (unsigned)(uintptr_t)&sW[0][srow * 32 + schunk],
      (unsigned)(uintptr_t)&sW[1][srow * 32 + schunk]};

  const __bf16* Arow0 = A + (long)(mBase + rlow) * Kpad + half * 8;
  const __bf16* Arow1 = A + (long)(mBase + 16 + rlow) * Kpad + half * 8;

  v8f c[2][4];
#pragma unroll
  for (int mt = 0; mt < 2; ++mt)
#pragma unroll
    for (int t = 0; t < 4; ++t) c[mt][t] = zero8f();

  int steps = Kpad >> 5;
  {  // prologue: stage k-step 0 into LDS buffer 0
    unsigned long long ga = (unsigned long long)(uintptr_t)wsrc;
    asm volatile("global_load_async_to_lds_b128 %0, %1, off"
                 :: "v"(ldsDst[0]), "v"(ga) : "memory");
  }
  v16bf a0 = load_tile_row(Arow0);
  v16bf a1 = load_tile_row(Arow1);

  for (int ks = 0; ks < steps; ++ks) {
    int cur = ks & 1;
    v16bf n0 = a0, n1 = a1;
    if (ks + 1 < steps) {
      // stage next W tile + load next A tiles while current k-step computes
      unsigned long long ga =
          (unsigned long long)(uintptr_t)(wsrc + (long)(ks + 1) * 32);
      asm volatile("global_load_async_to_lds_b128 %0, %1, off"
                   :: "v"(ldsDst[cur ^ 1]), "v"(ga) : "memory");
      n0 = load_tile_row(Arow0 + (ks + 1) * 32);
      n1 = load_tile_row(Arow1 + (ks + 1) * 32);
      asm volatile("s_wait_asynccnt 0x1" ::: "memory");  // current tile landed
    } else {
      asm volatile("s_wait_asynccnt 0x0" ::: "memory");
    }
    __syncthreads();  // whole 64x32 W tile visible to all waves

    // pull all 4 B tiles in one LDS clause, then run 8 WMMAs back-to-back
    v16bf bb[4];
#pragma unroll
    for (int t = 0; t < 4; ++t)
      bb[t] = load_tile_row(&sW[cur][(t * 16 + rlow) * 32 + half * 8]);
#pragma unroll
    for (int t = 0; t < 4; ++t) {
      c[0][t] = wmma_bf16(a0, bb[t], c[0][t]);
      c[1][t] = wmma_bf16(a1, bb[t], c[1][t]);
    }
    __syncthreads();  // all waves done reading before buffer reuse
    a0 = n0; a1 = n1;
  }

#pragma unroll
  for (int t = 0; t < 4; ++t) {
    int n = nBase + t * 16 + rlow;
    float bs = bias[n];
#pragma unroll
    for (int mt = 0; mt < 2; ++mt)
#pragma unroll
      for (int r = 0; r < 8; ++r) {
        int m = mBase + mt * 16 + r + half * 8;
        float v = fmaxf(c[mt][t][r] + bs, 0.0f);
        out[(long)m * ldo + n] = (__bf16)v;
      }
  }
}

// ---------- copy bottom-MLP output (Rpad cols 0..127) into T[:,0,:] ----------
__global__ void copy_x_to_T(const __bf16* __restrict__ Rpad, __bf16* __restrict__ T) {
  long i = (long)blockIdx.x * blockDim.x + threadIdx.x;
  if (i >= (long)BATCH * EMB_DIM) return;
  long b = i >> 7;
  int  k = (int)(i & 127);
  T[(long)b * 27 * EMB_DIM + k] = Rpad[b * 512 + k];
}

// ---------- zero Rpad padding columns 479..511 ----------
__global__ void zero_rpad_tail(__bf16* __restrict__ Rpad) {
  long i = (long)blockIdx.x * blockDim.x + threadIdx.x;
  if (i >= (long)BATCH * 33) return;
  long b = i / 33;
  int  c = 479 + (int)(i % 33);
  Rpad[b * 512 + c] = (__bf16)0.0f;
}

// ---------- interaction: per-sample Z = Tb . Tb^T, scatter strict-lower tri ----------
__global__ void __launch_bounds__(256) interact_wmma(
    const __bf16* __restrict__ T, __bf16* __restrict__ Rpad) {
  int lane = threadIdx.x & 31;
  int wave = threadIdx.x >> 5;
  int b = blockIdx.x * 8 + wave;
  int half = lane >> 4;
  int rlow = lane & 15;
  const __bf16* Tb = T + (long)b * 27 * EMB_DIM;

  int m1 = 16 + rlow;
  const __bf16* p0 = Tb + (long)rlow * EMB_DIM + half * 8;
  const __bf16* p1 = Tb + (long)m1 * EMB_DIM + half * 8;

  v8f c00 = zero8f(), c10 = zero8f(), c11 = zero8f();
#pragma unroll
  for (int ks = 0; ks < 4; ++ks) {
    int k0 = ks * 32;
    v16bf a0 = load_tile_row(p0 + k0);
    v16bf a1 = zero16bf();
    if (m1 < 27) a1 = load_tile_row(p1 + k0);
    c00 = wmma_bf16(a0, a0, c00);
    c10 = wmma_bf16(a1, a0, c10);
    c11 = wmma_bf16(a1, a1, c11);
  }

  __bf16* Rb = Rpad + (long)b * 512 + 128;
#pragma unroll
  for (int r = 0; r < 8; ++r) {
    int m, n;
    m = r + half * 8;      n = rlow;        // tile (0,0)
    if (m < 27 && n < m) Rb[m * (m - 1) / 2 + n] = (__bf16)c00[r];
    m = 16 + r + half * 8; n = rlow;        // tile (1,0)
    if (m < 27 && n < m) Rb[m * (m - 1) / 2 + n] = (__bf16)c10[r];
    m = 16 + r + half * 8; n = 16 + rlow;   // tile (1,1)
    if (m < 27 && n < m) Rb[m * (m - 1) / 2 + n] = (__bf16)c11[r];
  }
}

// ---------- final layer: 256 -> 1 dot + sigmoid ----------
__global__ void final_dot_sigmoid(const __bf16* __restrict__ act,
                                  const float* __restrict__ w,
                                  const float* __restrict__ bias,
                                  float* __restrict__ out) {
  int b = blockIdx.x * blockDim.x + threadIdx.x;
  if (b >= BATCH) return;
  const __bf16* a = act + (long)b * 256;
  float s = bias[0];
#pragma unroll 8
  for (int k = 0; k < 256; ++k) s += (float)a[k] * w[k];
  out[b] = 1.0f / (1.0f + __expf(-s));
}

extern "C" void kernel_launch(void* const* d_in, const int* in_sizes, int n_in,
                              void* d_out, int out_size, void* d_ws, size_t ws_size,
                              hipStream_t stream) {
  (void)in_sizes; (void)out_size;
  if (n_in < 19) return;
  // dict order: dense_x, lS_i, tables, bot_ws[0..2], bot_bs[0..2], top_ws[0..4], top_bs[0..4]
  const float* dense_x = (const float*)d_in[0];
  const int*   lSi     = (const int*)d_in[1];
  const float* tables  = (const float*)d_in[2];
  const float* bw[3] = {(const float*)d_in[3], (const float*)d_in[4], (const float*)d_in[5]};
  const float* bb[3] = {(const float*)d_in[6], (const float*)d_in[7], (const float*)d_in[8]};
  const float* tw[5] = {(const float*)d_in[9], (const float*)d_in[10], (const float*)d_in[11],
                        (const float*)d_in[12], (const float*)d_in[13]};
  const float* tb[5] = {(const float*)d_in[14], (const float*)d_in[15], (const float*)d_in[16],
                        (const float*)d_in[17], (const float*)d_in[18]};
  float* out = (float*)d_out;

  char* ws = (char*)d_ws;
  size_t off = 0;
  auto alloc = [&](size_t bytes) { size_t o = off; off = (off + bytes + 255) & ~(size_t)255; return o; };

  // bf16 padded weights: (N, Kpad)
  const int WN[7]   = {512, 256, 128, 1024, 1024, 512, 256};
  const int WK[7]   = {13, 512, 256, 479, 1024, 1024, 512};
  const int WKP[7]  = {32, 512, 256, 512, 1024, 1024, 512};
  size_t wOff[7];
  for (int i = 0; i < 7; ++i) wOff[i] = alloc((size_t)WN[i] * WKP[i] * 2);
  size_t xpadOff = alloc((size_t)BATCH * 32 * 2);
  size_t bufAOff = alloc((size_t)BATCH * 1024 * 2);
  size_t bufBOff = alloc((size_t)BATCH * 1024 * 2);
  size_t rpadOff = alloc((size_t)BATCH * 512 * 2);
  size_t tOff    = alloc((size_t)BATCH * 27 * EMB_DIM * 2);
  if (off > ws_size) return;

  __bf16* Wb[7];
  for (int i = 0; i < 7; ++i) Wb[i] = (__bf16*)(ws + wOff[i]);
  __bf16* Xpad = (__bf16*)(ws + xpadOff);
  __bf16* bufA = (__bf16*)(ws + bufAOff);
  __bf16* bufB = (__bf16*)(ws + bufBOff);
  __bf16* Rpad = (__bf16*)(ws + rpadOff);
  __bf16* Tbuf = (__bf16*)(ws + tOff);

  // --- convert weights (and padded dense input) to bf16 ---
  const float* wsrc[7] = {bw[0], bw[1], bw[2], tw[0], tw[1], tw[2], tw[3]};
  for (int i = 0; i < 7; ++i) {
    long tot = (long)WN[i] * WKP[i];
    cvt_pad_bf16<<<dim3((unsigned)((tot + 255) / 256)), 256, 0, stream>>>(
        wsrc[i], Wb[i], WK[i], WKP[i], tot);
  }
  {
    long tot = (long)BATCH * 32;
    cvt_pad_bf16<<<dim3((unsigned)((tot + 255) / 256)), 256, 0, stream>>>(
        dense_x, Xpad, 13, 32, tot);
  }

  auto gemm = [&](const __bf16* A, int Kpad, int wi, const float* bias,
                  __bf16* o, int N, int ldo) {
    dim3 g(N / 64, BATCH / 256);
    gemm_bf16_wmma<<<g, 256, 0, stream>>>(A, Wb[wi], bias, o, Kpad, ldo);
  };

  // --- bottom MLP ---
  gemm(Xpad, 32, 0, bb[0], bufA, 512, 512);
  gemm(bufA, 512, 1, bb[1], bufB, 256, 256);
  gemm(bufB, 256, 2, bb[2], Rpad, 128, 512);  // x -> Rpad cols 0..127

  // --- feature tensor T: x into slot 0, embeddings into slots 1..26 ---
  copy_x_to_T<<<(BATCH * EMB_DIM + 255) / 256, 256, 0, stream>>>(Rpad, Tbuf);
  embed_bag_sum<<<(NUM_TABLES * BATCH) / 8, 256, 0, stream>>>(lSi, tables, Tbuf);

  // --- interaction -> Rpad cols 128..478 ; zero 479..511 ---
  zero_rpad_tail<<<((long)BATCH * 33 + 255) / 256, 256, 0, stream>>>(Rpad);
  interact_wmma<<<BATCH / 8, 256, 0, stream>>>(Tbuf, Rpad);

  // --- top MLP ---
  gemm(Rpad, 512, 3, tb[0], bufA, 1024, 1024);
  gemm(bufA, 1024, 4, tb[1], bufB, 1024, 1024);
  gemm(bufB, 1024, 5, tb[2], bufA, 512, 512);
  gemm(bufA, 512, 6, tb[3], bufB, 256, 256);
  final_dot_sigmoid<<<BATCH / 256, 256, 0, stream>>>(bufB, tw[4], tb[4], out);
}
